// RelativeMultiHeadAttention_19224273617231
// MI455X (gfx1250) — matrix operand order repaired and checked by reading, not compiled, not run on hardware
//
#include <hip/hip_runtime.h>
#include <hip/hip_bf16.h>

// ---------------------------------------------------------------------------
// RelativeMultiHeadAttention for MI455X (gfx1250, wave32, WMMA f16->f32)
// B=2 S=2048 E=1024 H=16 D=64 R=128 P=257 RADIUS=128
// ---------------------------------------------------------------------------

typedef _Float16 h8  __attribute__((ext_vector_type(8)));
typedef _Float16 h16 __attribute__((ext_vector_type(16)));
typedef float    f8  __attribute__((ext_vector_type(8)));

#define S_LEN   2048
#define EMB     1024
#define NH      16
#define HD      64
#define P_REAL  257
#define P_PAD   272          // 17 tiles of 16
#define S_PAD   2320         // 128 ghost front + 2048 + 144 ghost back
#define NEG_BIG (-1.0e30f)

static __device__ __forceinline__ h16 concat8(h8 lo, h8 hi) {
  return __builtin_shufflevector(lo, hi, 0,1,2,3,4,5,6,7,8,9,10,11,12,13,14,15);
}
// A-frag (16x32 f16, row-major source): elems 0..7 = row[k0a+0..7], 8..15 = row[k0a+16..23]
static __device__ __forceinline__ h16 ldA(const _Float16* rp, int k0a) {
  h8 lo = *(const h8*)(rp + k0a);
  h8 hi = *(const h8*)(rp + k0a + 16);
  return concat8(lo, hi);
}
// B-frag (32x16 f16, B stored transposed [N][K]): elems 0..7 = row[k0b/2+0..7], 8..15 = row[k0b/2+16..23]
static __device__ __forceinline__ h16 ldB(const _Float16* rp, int k0b) {
  h8 lo = *(const h8*)(rp + (k0b >> 1));
  h8 hi = *(const h8*)(rp + (k0b >> 1) + 16);
  return concat8(lo, hi);
}
static __device__ __forceinline__ f8 wmma_step(h16 a, h16 b, f8 c) {
  return __builtin_amdgcn_wmma_f32_16x16x32_f16(false, a, false, b, (short)0, c, false, false);
}

// ------------------------------- conversions -------------------------------

__global__ void convert_qkv(const float* __restrict__ Q, const float* __restrict__ K,
                            const float* __restrict__ V,
                            _Float16* __restrict__ Qf, _Float16* __restrict__ Kf,
                            _Float16* __restrict__ Vf, int n) {
  int i = blockIdx.x * blockDim.x + threadIdx.x;
  if (i < n) {
    Qf[i] = (_Float16)Q[i];
    Kf[i] = (_Float16)K[i];
    Vf[i] = (_Float16)V[i];
  }
}

// WT[n][k] = W[k][n]  (f16, transposed so B-frags are contiguous)
__global__ void convert_w(const float* __restrict__ Wq, const float* __restrict__ Wk,
                          const float* __restrict__ Wv, const float* __restrict__ Wo,
                          _Float16* __restrict__ WqT, _Float16* __restrict__ WkT,
                          _Float16* __restrict__ WvT, _Float16* __restrict__ WoT) {
  int i = blockIdx.x * blockDim.x + threadIdx.x;
  if (i < EMB * EMB) {
    int nrow = i >> 10, k = i & 1023;
    int src = k * EMB + nrow;
    WqT[i] = (_Float16)Wq[src];
    WkT[i] = (_Float16)Wk[src];
    WvT[i] = (_Float16)Wv[src];
    WoT[i] = (_Float16)Wo[src];
  }
}

// EhT[h][p][d] = rel_table[p][d*H + h]   (rows 257..271 left as don't-care)
__global__ void convert_rel(const float* __restrict__ rel, _Float16* __restrict__ EhT) {
  int i = blockIdx.x * blockDim.x + threadIdx.x;
  if (i < NH * P_REAL * HD) {
    int h = i / (P_REAL * HD);
    int rem = i % (P_REAL * HD);
    int p = rem / HD, d = rem % HD;
    EhT[((size_t)h * P_PAD + p) * HD + d] = (_Float16)rel[(size_t)p * EMB + d * NH + h];
  }
}

// ------------------------- projection GEMMs (QKV) --------------------------
// z=0: Qh[b][h][s][d] = (X@Wq)/8 ; z=1: Kh padded ; z=2: VhT padded/transposed
__global__ void __launch_bounds__(32)
proj_kernel(const _Float16* __restrict__ Qf, const _Float16* __restrict__ Kf,
            const _Float16* __restrict__ Vf,
            const _Float16* __restrict__ WqT, const _Float16* __restrict__ WkT,
            const _Float16* __restrict__ WvT,
            _Float16* __restrict__ Qh, _Float16* __restrict__ Kh,
            _Float16* __restrict__ VhT) {
  int lane = threadIdx.x, ln = lane & 15, hiw = lane >> 4;
  int k0a = hiw ? 8 : 0, k0b = hiw ? 16 : 0;
  int m0 = blockIdx.x * 16, n0 = blockIdx.y * 16;
  int which = blockIdx.z;

  const _Float16* A  = (which == 0) ? Qf  : (which == 1) ? Kf  : Vf;
  const _Float16* BT = (which == 0) ? WqT : (which == 1) ? WkT : WvT;
  const _Float16* arow = A  + (size_t)(m0 + ln) * EMB;
  const _Float16* brow = BT + (size_t)(n0 + ln) * EMB;

  f8 c = {};
  for (int kk = 0; kk < EMB; kk += 32)
    c = wmma_step(ldA(arow + kk, k0a), ldB(brow + kk, k0b), c);

  int d = n0 >> 4;          // n0 multiple of 16 -> d fixed, head index = ln
#pragma unroll
  for (int r = 0; r < 8; ++r) {
    int m = m0 + r + 8 * hiw;
    int b = m >> 11, s = m & 2047;
    if (which == 0) {
      Qh[(((size_t)(b * NH + ln)) * S_LEN + s) * HD + d] = (_Float16)(c[r] * 0.125f);
    } else if (which == 1) {
      Kh[(((size_t)(b * NH + ln)) * S_PAD + s + 128) * HD + d] = (_Float16)c[r];
    } else {
      VhT[(((size_t)(b * NH + ln)) * HD + d) * S_PAD + (s + 128)] = (_Float16)c[r];
    }
  }
}

// ----------------------------- banded attention ----------------------------
__global__ void __launch_bounds__(32)
attn_kernel(const _Float16* __restrict__ Qh, const _Float16* __restrict__ Kh,
            const _Float16* __restrict__ VhT, const _Float16* __restrict__ EhT,
            _Float16* __restrict__ Ah) {
  __shared__ float qrel[16][P_PAD];
  __shared__ __align__(16) _Float16 plds[16][32];

  int lane = threadIdx.x, ln = lane & 15, hiw = lane >> 4;
  int k0a = hiw ? 8 : 0, k0b = hiw ? 16 : 0;
  int i0 = blockIdx.x * 16;
  int h = blockIdx.y, b = blockIdx.z;
  size_t bh = (size_t)b * NH + h;

  const _Float16* qrow = Qh + (bh * S_LEN + i0 + ln) * HD;  // already scaled by 1/8
  h16 aq0 = ldA(qrow, k0a);        // K(d) = 0..31
  h16 aq1 = ldA(qrow + 32, k0a);   // K(d) = 32..63

  // q_rel[q][p] = q . Eh[h][p]   (WMMA over D, 17 p-tiles)
  const _Float16* Ebase = EhT + bh % NH * (size_t)0 + (size_t)h * P_PAD * HD;
  for (int pt = 0; pt < 17; ++pt) {
    const _Float16* erow = Ebase + (size_t)(pt * 16 + ln) * HD;
    f8 acc = {};
    acc = wmma_step(aq0, ldB(erow, k0b), acc);
    acc = wmma_step(aq1, ldB(erow + 32, k0b), acc);
#pragma unroll
    for (int r = 0; r < 8; ++r) qrel[r + 8 * hiw][pt * 16 + ln] = acc[r];
  }
  __syncthreads();

  const _Float16* Kbase = Kh + bh * S_PAD * HD;
  const _Float16* Vbase = VhT + bh * HD * S_PAD;

  float Mr[8], Lr[8];
  f8 O[4] = {};
#pragma unroll
  for (int r = 0; r < 8; ++r) { Mr[r] = NEG_BIG; Lr[r] = 0.0f; }

  // band k in [i0-128, i0+159], 9 super-tiles of 32 columns
  for (int st = 0; st < 9; ++st) {
    int kb = i0 - 128 + st * 32;
    const _Float16* kr0 = Kbase + (size_t)(kb + 128 + ln) * HD;        // cols 0..15
    const _Float16* kr1 = kr0 + 16 * HD;                               // cols 16..31
    f8 e0 = {}, e1 = {};
    e0 = wmma_step(aq0, ldB(kr0, k0b), e0);
    e0 = wmma_step(aq1, ldB(kr0 + 32, k0b), e0);
    e1 = wmma_step(aq0, ldB(kr1, k0b), e1);
    e1 = wmma_step(aq1, ldB(kr1 + 32, k0b), e1);

    float p0[8], p1[8], sc[8];
#pragma unroll
    for (int r = 0; r < 8; ++r) {
      int m = r + 8 * hiw;
      int q = i0 + m;
      int kc0 = kb + ln, kc1 = kc0 + 16;
      int d0 = kc0 - q, d1 = kc1 - q;
      bool v0 = (d0 >= -128) && (d0 <= 128) && (kc0 >= 0) && (kc0 < S_LEN);
      bool v1 = (d1 >= -128) && (d1 <= 128) && (kc1 >= 0) && (kc1 < S_LEN);
      int g0 = min(max(d0 + 128, 0), 256);
      int g1 = min(max(d1 + 128, 0), 256);
      float s0 = v0 ? (e0[r] + qrel[m][g0]) : NEG_BIG;
      float s1 = v1 ? (e1[r] + qrel[m][g1]) : NEG_BIG;

      float mx = fmaxf(s0, s1);                 // row lives in one 16-lane half
      mx = fmaxf(mx, __shfl_xor(mx, 1, 32));
      mx = fmaxf(mx, __shfl_xor(mx, 2, 32));
      mx = fmaxf(mx, __shfl_xor(mx, 4, 32));
      mx = fmaxf(mx, __shfl_xor(mx, 8, 32));
      float nM = fmaxf(Mr[r], mx);
      float scale = __expf(Mr[r] - nM);
      p0[r] = __expf(s0 - nM);
      p1[r] = __expf(s1 - nM);
      float rs = p0[r] + p1[r];
      rs += __shfl_xor(rs, 1, 32);
      rs += __shfl_xor(rs, 2, 32);
      rs += __shfl_xor(rs, 4, 32);
      rs += __shfl_xor(rs, 8, 32);
      Lr[r] = Lr[r] * scale + rs;
      Mr[r] = nM;
      sc[r] = scale;
    }

    // C/D layout -> A-frag layout via LDS
#pragma unroll
    for (int r = 0; r < 8; ++r) {
      plds[r + 8 * hiw][ln]      = (_Float16)p0[r];
      plds[r + 8 * hiw][ln + 16] = (_Float16)p1[r];
    }
    __syncthreads();
    h16 pa = ldA(&plds[ln][0], k0a);

#pragma unroll
    for (int dt = 0; dt < 4; ++dt) {
      const _Float16* vrow = Vbase + (size_t)(dt * 16 + ln) * S_PAD + (kb + 128);
      f8 o = O[dt];
#pragma unroll
      for (int r = 0; r < 8; ++r) o[r] *= sc[r];
      O[dt] = wmma_step(pa, ldB(vrow, k0b), o);
    }
    __syncthreads();
  }

#pragma unroll
  for (int r = 0; r < 8; ++r) Lr[r] = 1.0f / Lr[r];
#pragma unroll
  for (int dt = 0; dt < 4; ++dt) {
#pragma unroll
    for (int r = 0; r < 8; ++r) {
      int m = r + 8 * hiw;
      int s = i0 + m;
      int d = dt * 16 + ln;
      Ah[(((size_t)b * S_LEN + s)) * EMB + d * NH + h] = (_Float16)(O[dt][r] * Lr[r]);
    }
  }
}

// --------------------------- output projection -----------------------------
__global__ void __launch_bounds__(32)
out_gemm(const _Float16* __restrict__ Ah, const _Float16* __restrict__ WoT,
         float* __restrict__ out) {
  int lane = threadIdx.x, ln = lane & 15, hiw = lane >> 4;
  int k0a = hiw ? 8 : 0, k0b = hiw ? 16 : 0;
  int m0 = blockIdx.x * 16, n0 = blockIdx.y * 16;
  const _Float16* arow = Ah  + (size_t)(m0 + ln) * EMB;
  const _Float16* brow = WoT + (size_t)(n0 + ln) * EMB;
  f8 c = {};
  for (int kk = 0; kk < EMB; kk += 32)
    c = wmma_step(ldA(arow + kk, k0a), ldB(brow + kk, k0b), c);
#pragma unroll
  for (int r = 0; r < 8; ++r)
    out[(size_t)(m0 + r + 8 * hiw) * EMB + n0 + ln] = c[r];
}

// --------------------------------- launch ----------------------------------
extern "C" void kernel_launch(void* const* d_in, const int* in_sizes, int n_in,
                              void* d_out, int out_size, void* d_ws, size_t ws_size,
                              hipStream_t stream) {
  const float* Q   = (const float*)d_in[0];
  const float* K   = (const float*)d_in[1];
  const float* V   = (const float*)d_in[2];
  const float* Wq  = (const float*)d_in[3];
  const float* Wk  = (const float*)d_in[4];
  const float* Wv  = (const float*)d_in[5];
  const float* Wo  = (const float*)d_in[6];
  const float* rel = (const float*)d_in[7];

  char* w = (char*)d_ws;
  const size_t MROWS = (size_t)2 * S_LEN;         // 4096
  _Float16* Qf  = (_Float16*)w; w += MROWS * EMB * 2;
  _Float16* Kf  = (_Float16*)w; w += MROWS * EMB * 2;
  _Float16* Vf  = (_Float16*)w; w += MROWS * EMB * 2;
  _Float16* WqT = (_Float16*)w; w += (size_t)EMB * EMB * 2;
  _Float16* WkT = (_Float16*)w; w += (size_t)EMB * EMB * 2;
  _Float16* WvT = (_Float16*)w; w += (size_t)EMB * EMB * 2;
  _Float16* WoT = (_Float16*)w; w += (size_t)EMB * EMB * 2;
  _Float16* EhT = (_Float16*)w; w += (size_t)NH * P_PAD * HD * 2;
  _Float16* Qh  = (_Float16*)w; w += (size_t)2 * NH * S_LEN * HD * 2;
  _Float16* Kh  = (_Float16*)w; w += (size_t)2 * NH * S_PAD * HD * 2;
  _Float16* VhT = (_Float16*)w; w += (size_t)2 * NH * HD * S_PAD * 2;
  _Float16* Ah  = (_Float16*)w; w += MROWS * EMB * 2;

  int nqkv = (int)(MROWS * EMB);
  convert_qkv<<<(nqkv + 255) / 256, 256, 0, stream>>>(Q, K, V, Qf, Kf, Vf, nqkv);
  convert_w<<<(EMB * EMB + 255) / 256, 256, 0, stream>>>(Wq, Wk, Wv, Wo, WqT, WkT, WvT, WoT);
  int nrel = NH * P_REAL * HD;
  convert_rel<<<(nrel + 255) / 256, 256, 0, stream>>>(rel, EhT);

  proj_kernel<<<dim3(MROWS / 16, EMB / 16, 3), 32, 0, stream>>>(
      Qf, Kf, Vf, WqT, WkT, WvT, Qh, Kh, VhT);

  attn_kernel<<<dim3(S_LEN / 16, NH, 2), 32, 0, stream>>>(Qh, Kh, VhT, EhT, Ah);

  out_gemm<<<dim3(MROWS / 16, EMB / 16), 32, 0, stream>>>(Ah, WoT, (float*)d_out);
}
